// NeuroSAT_75385265979966
// MI455X (gfx1250) — matrix operand order repaired and checked by reading, not compiled, not run on hardware
//
#include <hip/hip_runtime.h>
#include <hip/hip_bf16.h>

// ---------------------------------------------------------------------------
// NeuroSAT message passing for MI455X (gfx1250, wave32, WMMA).
// Dense layer GEMMs -> v_wmma_f32_16x16x32_bf16 ; adjacency matmuls -> sparse
// CSR gathers (density 0.0015 => ~6 nnz/clause; dense would waste 600x FLOPs).
// ---------------------------------------------------------------------------

#define DIM       128
#define N_VARS    2048
#define N_LITS    4096
#define N_CLAUSES 8192
#define N_ROUNDS  26
#define MAXC      64    // cap nnz per clause column (Poisson mean ~6.1)
#define MAXL      96    // cap nnz per literal row  (Poisson mean ~12.3)

typedef __attribute__((ext_vector_type(16))) __bf16          v16bf;
typedef __attribute__((ext_vector_type(8)))  float           v8f;
typedef __attribute__((ext_vector_type(16))) unsigned short  v16us;
typedef __attribute__((ext_vector_type(8)))  unsigned short  v8us;

// ---- helpers --------------------------------------------------------------

__device__ __forceinline__ unsigned short f2bf(float x) {
    unsigned u = __builtin_bit_cast(unsigned, x);
    u += 0x7FFFu + ((u >> 16) & 1u);          // round-to-nearest-even
    return (unsigned short)(u >> 16);
}

__device__ __forceinline__ float sigm(float x) {
    return 1.0f / (1.0f + __expf(-x));
}

// A tile: 16x32 bf16 from row-major bf16 buffer (LDS), leading dim strideE.
// Lane m (0-15) holds row M=m, K = {0..15}; lanes 16-31 hold K={16..31}.
__device__ __forceinline__ v16bf load_a_tile(const unsigned short* buf,
                                             int strideE, int k0, int lane) {
    const int m  = lane & 15;
    const int kb = (lane >> 4) << 4;          // 0 or 16
    const unsigned short* p = buf + m * strideE + k0 + kb;
    v8us lo = *(const v8us*)p;
    v8us hi = *(const v8us*)(p + 8);
    v16us a;
#pragma unroll
    for (int i = 0; i < 8; ++i) { a[i] = lo[i]; a[i + 8] = hi[i]; }
    return __builtin_bit_cast(v16bf, a);
}

// B tile: 32x16 (KxN) from row-major bf16 weight W[N][K] (out = x @ W^T).
// Lane L holds column N = L&15, 16 contiguous K starting at k0 + 16*(L>>4).
__device__ __forceinline__ v16bf load_b_tile(const unsigned short* __restrict__ W,
                                             int K, int n0, int k0, int lane) {
    const int n  = n0 + (lane & 15);
    const int kb = k0 + ((lane >> 4) << 4);
    v16us b = *(const v16us*)(W + (size_t)n * K + kb);
    return __builtin_bit_cast(v16bf, b);
}

// ---- one-time setup kernels ----------------------------------------------

__global__ __launch_bounds__(256) void convert_weights_kernel(
    const float* __restrict__ lmsg_w1, const float* __restrict__ lmsg_w2,
    const float* __restrict__ lmsg_w3, const float* __restrict__ cmsg_w1,
    const float* __restrict__ cmsg_w2, const float* __restrict__ cmsg_w3,
    const float* __restrict__ lvote_w1, const float* __restrict__ lvote_w2,
    const float* __restrict__ c_wih, const float* __restrict__ c_whh,
    const float* __restrict__ c_bih, const float* __restrict__ c_bhh,
    const float* __restrict__ l_wih, const float* __restrict__ l_whh,
    const float* __restrict__ l_bih, const float* __restrict__ l_bhh,
    unsigned short* __restrict__ bw,      // 8 x 16384 bf16 matrices
    unsigned short* __restrict__ ccat,    // [512][256] bf16
    unsigned short* __restrict__ lcat,    // [512][384] bf16
    float* __restrict__ cb, float* __restrict__ lb)
{
    const int i = blockIdx.x * blockDim.x + threadIdx.x;
    const int S8 = 8 * 16384;
    const int SC = 512 * 256;
    const int SL = 512 * 384;
    if (i < S8) {
        const int m = i >> 14, j = i & 16383;
        const float* src;
        switch (m) {
            case 0: src = lmsg_w1; break;  case 1: src = lmsg_w2; break;
            case 2: src = lmsg_w3; break;  case 3: src = cmsg_w1; break;
            case 4: src = cmsg_w2; break;  case 5: src = cmsg_w3; break;
            case 6: src = lvote_w1; break; default: src = lvote_w2; break;
        }
        bw[i] = f2bf(src[j]);
    } else if (i < S8 + SC) {
        const int j = i - S8, n = j >> 8, k = j & 255;
        const float v = (k < 128) ? c_wih[n * 128 + k] : c_whh[n * 128 + (k - 128)];
        ccat[j] = f2bf(v);
    } else if (i < S8 + SC + SL) {
        const int j = i - S8 - SC, n = j / 384, k = j % 384;
        const float v = (k < 256) ? l_wih[n * 256 + k] : l_whh[n * 128 + (k - 256)];
        lcat[j] = f2bf(v);
    } else if (i < S8 + SC + SL + 512) {
        const int j = i - S8 - SC - SL;
        cb[j] = c_bih[j] + c_bhh[j];
    } else if (i < S8 + SC + SL + 1024) {
        const int j = i - S8 - SC - SL - 512;
        lb[j] = l_bih[j] + l_bhh[j];
    }
}

// Per-clause CSR: thread c scans column c (coalesced across lanes).
__global__ __launch_bounds__(256) void build_clause_csr(
    const float* __restrict__ adj, int* __restrict__ idx, int* __restrict__ cnt)
{
    const int c = blockIdx.x * blockDim.x + threadIdx.x;
    if (c >= N_CLAUSES) return;
    int n = 0;
    for (int l = 0; l < N_LITS; ++l) {
        const float v = adj[(size_t)l * N_CLAUSES + c];
        if (v != 0.0f && n < MAXC) idx[(size_t)c * MAXC + n++] = l;
    }
    cnt[c] = n;
}

// Per-literal CSR: thread l streams its contiguous row (float4 vectorized).
__global__ __launch_bounds__(256) void build_lit_csr(
    const float* __restrict__ adj, int* __restrict__ idx, int* __restrict__ cnt)
{
    const int l = blockIdx.x * blockDim.x + threadIdx.x;
    if (l >= N_LITS) return;
    int n = 0;
    const float4* row = (const float4*)(adj + (size_t)l * N_CLAUSES);
    for (int c4 = 0; c4 < N_CLAUSES / 4; ++c4) {
        const float4 v = row[c4];
        const int c = c4 * 4;
        if (v.x != 0.0f && n < MAXL) idx[(size_t)l * MAXL + n++] = c;
        if (v.y != 0.0f && n < MAXL) idx[(size_t)l * MAXL + n++] = c + 1;
        if (v.z != 0.0f && n < MAXL) idx[(size_t)l * MAXL + n++] = c + 2;
        if (v.w != 0.0f && n < MAXL) idx[(size_t)l * MAXL + n++] = c + 3;
    }
    cnt[l] = n;
}

__global__ __launch_bounds__(256) void init_states_kernel(
    const float* __restrict__ lw, const float* __restrict__ lbi,
    const float* __restrict__ cw, const float* __restrict__ cbi,
    float* __restrict__ Lh, float* __restrict__ Lc,
    float* __restrict__ Ch, float* __restrict__ Cc)
{
    const int i = blockIdx.x * blockDim.x + threadIdx.x; // (N_LITS+N_CLAUSES)*128
    const int d = i & 127, row = i >> 7;
    if (row < N_LITS) {
        Lh[i] = lw[d] + lbi[d];
        Lc[i] = 0.0f;
    } else {
        const int j = (row - N_LITS) * DIM + d;
        Ch[j] = cw[d] + cbi[d];
        Cc[j] = 0.0f;
    }
}

// ---- fused MLP (2 or 3 layers of 128x128), bf16 WMMA ---------------------
// Block = 128 threads (4 waves) processing 16 rows; activations ping-pong in
// LDS (stride 136 bf16 to break bank conflicts). Weights stream from L2.

template <bool THREE>
__global__ __launch_bounds__(128) void mlp_kernel(
    const float* __restrict__ X,
    const unsigned short* __restrict__ W1, const unsigned short* __restrict__ W2,
    const unsigned short* __restrict__ W3,
    const float* __restrict__ B1, const float* __restrict__ B2,
    const float* __restrict__ B3,
    float* __restrict__ Y)
{
    __shared__ alignas(16) unsigned short actA[16 * 136];
    __shared__ alignas(16) unsigned short actB[16 * 136];
    const int tid  = threadIdx.x;
    const int lane = tid & 31;
    const int wave = tid >> 5;
    const int row0 = blockIdx.x * 16;

#pragma unroll
    for (int r = 0; r < 16; ++r)
        actA[r * 136 + tid] = f2bf(X[(size_t)(row0 + r) * DIM + tid]);
    __syncthreads();

    unsigned short* src = actA;
    unsigned short* dst = actB;
    const int NL = THREE ? 3 : 2;
    for (int l = 0; l < NL; ++l) {
        const unsigned short* W  = (l == 0) ? W1 : (l == 1 ? W2 : W3);
        const float*          Bv = (l == 0) ? B1 : (l == 1 ? B2 : B3);
        const bool relu = THREE ? (l < 2) : true;
        const bool last = (l == NL - 1);
#pragma unroll
        for (int t = 0; t < 2; ++t) {
            const int n0 = (wave * 2 + t) * 16;
            v8f acc = {};
#pragma unroll
            for (int k0 = 0; k0 < DIM; k0 += 32) {
                v16bf a = load_a_tile(src, 136, k0, lane);
                v16bf b = load_b_tile(W, DIM, n0, k0, lane);
                acc = __builtin_amdgcn_wmma_f32_16x16x32_bf16(
                    false, a, false, b, (short)0, acc, false, false);
            }
            const int   n    = n0 + (lane & 15);
            const int   mh   = (lane >> 4) * 8;
            const float bias = Bv[n];
#pragma unroll
            for (int r = 0; r < 8; ++r) {
                float v = acc[r] + bias;
                if (relu) v = v > 0.0f ? v : 0.0f;
                if (last) Y[(size_t)(row0 + mh + r) * DIM + n] = v;
                else      dst[(mh + r) * 136 + n] = f2bf(v);
            }
        }
        __syncthreads();
        unsigned short* tmp = src; src = dst; dst = tmp;
    }
}

// ---- fused LSTM step: gates = [x|h] @ Wcat^T + b, then elementwise -------
// K = 256 (clause LSTM) or 384 (literal LSTM with flip concat).

__global__ __launch_bounds__(128) void lstm_kernel(
    const float* __restrict__ Xa,    // [M,128] message input
    const float* __restrict__ Flip,  // Lh for flip part (K==384), else unused
    const float* __restrict__ Hin,   // [M,128] hidden input
    const unsigned short* __restrict__ Wcat,  // [512][K] bf16
    const float* __restrict__ Bcat,  // [512] bih+bhh
    const float* __restrict__ Cin,
    float* __restrict__ Hout, float* __restrict__ Cout,
    int M, int K)
{
    __shared__ alignas(16) unsigned short acts[16 * 392];
    __shared__ float gates[16 * 512];
    const int tid  = threadIdx.x;
    const int lane = tid & 31;
    const int wave = tid >> 5;
    const int row0 = blockIdx.x * 16;
    const int strideE = K + 8;
    const int Mh = M >> 1;

    for (int r = 0; r < 16; ++r) {
        const int gr = row0 + r;
        for (int c = tid; c < K; c += 128) {
            float v;
            if (c < 128)        v = Xa[(size_t)gr * DIM + c];
            else if (K == 256)  v = Hin[(size_t)gr * DIM + (c - 128)];
            else if (c < 256)   v = Flip[(size_t)((gr + Mh) % M) * DIM + (c - 128)];
            else                v = Hin[(size_t)gr * DIM + (c - 256)];
            acts[r * strideE + c] = f2bf(v);
        }
    }
    __syncthreads();

    for (int t = 0; t < 8; ++t) {                 // 4 waves x 8 tiles = 512 gate cols
        const int n0 = (wave * 8 + t) * 16;
        v8f acc = {};
        for (int k0 = 0; k0 < K; k0 += 32) {
            v16bf a = load_a_tile(acts, strideE, k0, lane);
            v16bf b = load_b_tile(Wcat, K, n0, k0, lane);
            acc = __builtin_amdgcn_wmma_f32_16x16x32_bf16(
                false, a, false, b, (short)0, acc, false, false);
        }
        const int   n    = n0 + (lane & 15);
        const int   mh   = (lane >> 4) * 8;
        const float bias = Bcat[n];
#pragma unroll
        for (int r = 0; r < 8; ++r)
            gates[(mh + r) * 512 + n] = acc[r] + bias;
    }
    __syncthreads();

#pragma unroll
    for (int r = 0; r < 16; ++r) {
        const int d = tid;
        const float* g = gates + r * 512;
        const float ig = sigm(g[d]);
        const float fg = sigm(g[128 + d]);
        const float gg = tanhf(g[256 + d]);
        const float og = sigm(g[384 + d]);
        const size_t o = (size_t)(row0 + r) * DIM + d;
        const float cn = fg * Cin[o] + ig * gg;
        Cout[o] = cn;
        Hout[o] = og * tanhf(cn);
    }
}

// ---- sparse gather: dst[g,:] = sum over CSR rows of src ------------------

__global__ __launch_bounds__(256) void gather_kernel(
    const float* __restrict__ src, const int* __restrict__ idx,
    const int* __restrict__ cnt, int maxnnz,
    float* __restrict__ dst, int Ndst)
{
    const int g = blockIdx.x * 2 + (threadIdx.x >> 7);
    const int d = threadIdx.x & 127;
    if (g >= Ndst) return;
    const int n = cnt[g];
    const int* ip = idx + (size_t)g * maxnnz;
    float s = 0.0f;
    for (int j = 0; j < n; ++j)
        s += src[(size_t)ip[j] * DIM + d];
    dst[(size_t)g * DIM + d] = s;
}

// ---- vote tail -----------------------------------------------------------

__global__ __launch_bounds__(256) void vote_dot_kernel(
    const float* __restrict__ H, const float* __restrict__ w3,
    const float* __restrict__ b3, float* __restrict__ vote)
{
    const int r = blockIdx.x * blockDim.x + threadIdx.x;
    if (r >= N_LITS) return;
    float s = b3[0];
    for (int k = 0; k < DIM; ++k)
        s += H[(size_t)r * DIM + k] * w3[k];
    vote[r] = s;
}

__global__ __launch_bounds__(256) void reduce_mean_kernel(
    const float* __restrict__ vote, float* __restrict__ out0)
{
    __shared__ float s[256];
    const int t = threadIdx.x;
    float a = 0.0f;
    for (int i = t; i < N_LITS; i += 256) a += vote[i];
    s[t] = a;
    __syncthreads();
    for (int w = 128; w > 0; w >>= 1) {
        if (t < w) s[t] += s[t + w];
        __syncthreads();
    }
    if (t == 0) out0[0] = s[0] / (float)N_LITS;
}

__global__ __launch_bounds__(256) void copy_kernel(
    const float* __restrict__ src, float* __restrict__ dst, int n)
{
    const int i = blockIdx.x * blockDim.x + threadIdx.x;
    if (i < n) dst[i] = src[i];
}

// ---------------------------------------------------------------------------

extern "C" void kernel_launch(void* const* d_in, const int* in_sizes, int n_in,
                              void* d_out, int out_size, void* d_ws, size_t ws_size,
                              hipStream_t stream) {
    (void)in_sizes; (void)n_in; (void)out_size; (void)ws_size;

    const float* adj      = (const float*)d_in[0];
    const float* l_init_w = (const float*)d_in[1];
    const float* l_init_b = (const float*)d_in[2];
    const float* c_init_w = (const float*)d_in[3];
    const float* c_init_b = (const float*)d_in[4];
    const float* lmsg_w1  = (const float*)d_in[5];
    const float* lmsg_b1  = (const float*)d_in[6];
    const float* lmsg_w2  = (const float*)d_in[7];
    const float* lmsg_b2  = (const float*)d_in[8];
    const float* lmsg_w3  = (const float*)d_in[9];
    const float* lmsg_b3  = (const float*)d_in[10];
    const float* cmsg_w1  = (const float*)d_in[11];
    const float* cmsg_b1  = (const float*)d_in[12];
    const float* cmsg_w2  = (const float*)d_in[13];
    const float* cmsg_b2  = (const float*)d_in[14];
    const float* cmsg_w3  = (const float*)d_in[15];
    const float* cmsg_b3  = (const float*)d_in[16];
    const float* l_wih    = (const float*)d_in[17];
    const float* l_whh    = (const float*)d_in[18];
    const float* l_bih    = (const float*)d_in[19];
    const float* l_bhh    = (const float*)d_in[20];
    const float* c_wih    = (const float*)d_in[21];
    const float* c_whh    = (const float*)d_in[22];
    const float* c_bih    = (const float*)d_in[23];
    const float* c_bhh    = (const float*)d_in[24];
    const float* lvote_w1 = (const float*)d_in[25];
    const float* lvote_b1 = (const float*)d_in[26];
    const float* lvote_w2 = (const float*)d_in[27];
    const float* lvote_b2 = (const float*)d_in[28];
    const float* lvote_w3 = (const float*)d_in[29];
    const float* lvote_b3 = (const float*)d_in[30];

    // --- workspace layout (256B-aligned slabs) ---
    char* ws = (char*)d_ws;
    size_t off = 0;
    auto alloc = [&](size_t bytes) {
        size_t o = off;
        off += (bytes + 255) & ~(size_t)255;
        return ws + o;
    };
    float* Lh0   = (float*)alloc((size_t)N_LITS * DIM * 4);
    float* Lh1   = (float*)alloc((size_t)N_LITS * DIM * 4);
    float* Lc    = (float*)alloc((size_t)N_LITS * DIM * 4);
    float* Ch    = (float*)alloc((size_t)N_CLAUSES * DIM * 4);
    float* Cc    = (float*)alloc((size_t)N_CLAUSES * DIM * 4);
    float* Lpre  = (float*)alloc((size_t)N_LITS * DIM * 4);
    float* Cpre  = (float*)alloc((size_t)N_CLAUSES * DIM * 4);
    float* LCmsg = (float*)alloc((size_t)N_CLAUSES * DIM * 4);
    float* CLmsg = (float*)alloc((size_t)N_LITS * DIM * 4);
    float* Hvote = (float*)alloc((size_t)N_LITS * DIM * 4);
    unsigned short* bw   = (unsigned short*)alloc((size_t)8 * 16384 * 2);
    unsigned short* ccat = (unsigned short*)alloc((size_t)512 * 256 * 2);
    unsigned short* lcat = (unsigned short*)alloc((size_t)512 * 384 * 2);
    float* cbB = (float*)alloc(512 * 4);
    float* lbB = (float*)alloc(512 * 4);
    int* cidx = (int*)alloc((size_t)N_CLAUSES * MAXC * 4);
    int* ccnt = (int*)alloc((size_t)N_CLAUSES * 4);
    int* lidx = (int*)alloc((size_t)N_LITS * MAXL * 4);
    int* lcnt = (int*)alloc((size_t)N_LITS * 4);

    unsigned short* bw_lmsg1 = bw + 0 * 16384;
    unsigned short* bw_lmsg2 = bw + 1 * 16384;
    unsigned short* bw_lmsg3 = bw + 2 * 16384;
    unsigned short* bw_cmsg1 = bw + 3 * 16384;
    unsigned short* bw_cmsg2 = bw + 4 * 16384;
    unsigned short* bw_cmsg3 = bw + 5 * 16384;
    unsigned short* bw_lv1   = bw + 6 * 16384;
    unsigned short* bw_lv2   = bw + 7 * 16384;

    // --- one-time setup ---
    {
        const int total = 8 * 16384 + 512 * 256 + 512 * 384 + 1024;
        convert_weights_kernel<<<(total + 255) / 256, 256, 0, stream>>>(
            lmsg_w1, lmsg_w2, lmsg_w3, cmsg_w1, cmsg_w2, cmsg_w3, lvote_w1,
            lvote_w2, c_wih, c_whh, c_bih, c_bhh, l_wih, l_whh, l_bih, l_bhh,
            bw, ccat, lcat, cbB, lbB);
    }
    build_clause_csr<<<N_CLAUSES / 256, 256, 0, stream>>>(adj, cidx, ccnt);
    build_lit_csr<<<N_LITS / 256, 256, 0, stream>>>(adj, lidx, lcnt);
    init_states_kernel<<<(N_LITS + N_CLAUSES) * DIM / 256, 256, 0, stream>>>(
        l_init_w, l_init_b, c_init_w, c_init_b, Lh0, Lc, Ch, Cc);

    // --- 26 message-passing rounds ---
    for (int r = 0; r < N_ROUNDS; ++r) {
        float* Lcur = (r & 1) ? Lh1 : Lh0;
        float* Lnxt = (r & 1) ? Lh0 : Lh1;

        mlp_kernel<true><<<N_LITS / 16, 128, 0, stream>>>(
            Lcur, bw_lmsg1, bw_lmsg2, bw_lmsg3, lmsg_b1, lmsg_b2, lmsg_b3, Lpre);

        gather_kernel<<<N_CLAUSES / 2, 256, 0, stream>>>(
            Lpre, cidx, ccnt, MAXC, LCmsg, N_CLAUSES);

        lstm_kernel<<<N_CLAUSES / 16, 128, 0, stream>>>(
            LCmsg, nullptr, Ch, ccat, cbB, Cc, Ch, Cc, N_CLAUSES, 256);

        mlp_kernel<true><<<N_CLAUSES / 16, 128, 0, stream>>>(
            Ch, bw_cmsg1, bw_cmsg2, bw_cmsg3, cmsg_b1, cmsg_b2, cmsg_b3, Cpre);

        gather_kernel<<<N_LITS / 2, 256, 0, stream>>>(
            Cpre, lidx, lcnt, MAXL, CLmsg, N_LITS);

        lstm_kernel<<<N_LITS / 16, 128, 0, stream>>>(
            CLmsg, Lcur, Lcur, lcat, lbB, Lc, Lnxt, Lc, N_LITS, 384);
    }
    // after 26 rounds, final Lh is in Lh0

    // --- vote head: 2-layer relu MLP (WMMA) + 1-wide dot + mean ---
    float* out = (float*)d_out;
    mlp_kernel<false><<<N_LITS / 16, 128, 0, stream>>>(
        Lh0, bw_lv1, bw_lv2, nullptr, lvote_b1, lvote_b2, nullptr, Hvote);
    vote_dot_kernel<<<N_LITS / 256, 256, 0, stream>>>(Hvote, lvote_w3, lvote_b3,
                                                      out + 1);
    reduce_mean_kernel<<<1, 256, 0, stream>>>(out + 1, out);
    copy_kernel<<<(N_LITS * DIM) / 256, 256, 0, stream>>>(
        Lh0, out + 1 + N_LITS, N_LITS * DIM);
}
